// LSTMLayer_34900904247945
// MI455X (gfx1250) — matrix operand order repaired and checked by reading, not compiled, not run on hardware
//
#include <hip/hip_runtime.h>
#include <hip/hip_bf16.h>

typedef __attribute__((ext_vector_type(16))) _Float16 v16h;
typedef __attribute__((ext_vector_type(8)))  float    v8f;

#define TSTEPS 512
#define BB     64
#define INS    256
#define HH     512
#define KTOT   768          // 256 (x) + 512 (h)
#define NKT    24           // KTOT / 32
#define NGATE  4
#define NCOLS  2048         // 4 gates * 512

struct Ptrs {
    const float* wx[4];   // [H, IN] row-major, gate order i,j,f,o
    const float* wh[4];   // [H, H]
    const float* bx[4];   // [H]
    const float* bh[4];   // [H]
};

// ---------------------------------------------------------------------------
// Prep: pack [Wx;Wh] (768 x 2048) into f16 WMMA B-fragment layout, fold biases.
// B tile (32x16, f16): lane = n%16 + 16*(k_in_tile/16), value idx = k_in_tile%16
// Blob layout: Bp[(ntile*24 + ktile)*512 + lane*16 + idx]
// ---------------------------------------------------------------------------
__global__ __launch_bounds__(256) void prep_pack(Ptrs P, _Float16* __restrict__ Bp,
                                                 float* __restrict__ biascat) {
    int id = blockIdx.x * 256 + threadIdx.x;
    if (id < NCOLS) {
        int g = id >> 9, j = id & 511;
        biascat[id] = P.bx[g][j] + P.bh[g][j];
    }
    if (id >= KTOT * NCOLS) return;
    int n = id / KTOT;                 // output column (gate-major)
    int k = id % KTOT;                 // reduction index
    int g = n >> 9, j = n & 511;
    float v = (k < INS) ? P.wx[g][j * INS + k] : P.wh[g][j * HH + (k - INS)];
    int ntile = n >> 4, nn = n & 15;
    int kt = k >> 5,  kq = k & 31;
    int lane = nn + ((kq >> 4) << 4);
    int idx  = kq & 15;
    Bp[(size_t)(ntile * NKT + kt) * 512 + lane * 16 + idx] = (_Float16)v;
}

// ---------------------------------------------------------------------------
// One LSTM time step. Grid = 16 WGs (WG wg owns h-cols [32wg,32wg+32) for all
// 4 gates), 256 threads = 8 waves. Wave w: gate g=w/2, n-tile = g*32+wg*2+(w&1),
// all 4 m-tiles (64 batch rows).
// ---------------------------------------------------------------------------
__global__ __launch_bounds__(256) void lstm_step(
    const float* __restrict__ x_t,     // [64, 256]
    const float* __restrict__ h_src,   // [64, 512]
    const float* __restrict__ c_src,   // [64, 512]
    float*       __restrict__ c_dst,   // [64, 512] (may alias c_src)
    const _Float16* __restrict__ Bp,   // packed weights
    const float* __restrict__ biascat, // [2048]
    float*       __restrict__ h_out)   // [64, 512] -> d_out slice for step t
{
    __shared__ __align__(32) unsigned char smem[32768];
    _Float16* ldsA = (_Float16*)smem;                   // A fragments: 4 mtiles x 8 ktiles x 512
    float (*ldsP)[BB][32] = (float(*)[BB][32])smem;     // gate exchange (reused after GEMM)

    const int tid  = threadIdx.x;
    const int lane = tid & 31;
    const int w    = tid >> 5;
    const int wg   = blockIdx.x;                        // 0..15
    const int gate = w >> 1;
    const int ntile = gate * 32 + wg * 2 + (w & 1);

    v8f acc[4];
    #pragma unroll
    for (int m = 0; m < 4; ++m)
        #pragma unroll
        for (int p = 0; p < 8; ++p) acc[m][p] = 0.0f;

    // K processed in 3 chunks of 8 k-tiles (256 K each) to keep LDS at 32KB.
    for (int chunk = 0; chunk < 3; ++chunk) {
        // ---- stage A = [x_t | h] chunk into LDS in WMMA A-fragment layout ----
        // 64 rows x 128 f16-pairs per chunk = 8192 pairs
        for (int p = tid; p < 8192; p += 256) {
            int b  = p >> 7;
            int pk = p & 127;
            int k  = chunk * 256 + pk * 2;
            float v0, v1;
            if (k < INS) { v0 = x_t[b * INS + k];        v1 = x_t[b * INS + k + 1]; }
            else         { v0 = h_src[b * HH + (k-INS)]; v1 = h_src[b * HH + (k-INS) + 1]; }
            int mtile = b >> 4, r = b & 15;
            int kt  = (k >> 5) - chunk * 8;              // 0..7
            int kq  = k & 31;
            int ln  = r + ((kq >> 4) << 4);              // lanes 0-15: K 0-15; 16-31: K 16-31
            int idx = kq & 15;
            int off = (mtile * 8 + kt) * 512 + ln * 16 + idx;
            ldsA[off]     = (_Float16)v0;
            ldsA[off + 1] = (_Float16)v1;
        }
        __syncthreads();

        // ---- GEMM: 8 k-tiles x 4 m-tiles of v_wmma_f32_16x16x32_f16 ----
        const _Float16* bbase = Bp + (size_t)(ntile * NKT + chunk * 8) * 512 + lane * 16;
        if (chunk < 2) __builtin_prefetch(bbase + 8 * 512, 0, 1);   // next chunk's weights
        #pragma unroll
        for (int kt = 0; kt < 8; ++kt) {
            v16h bfrag = *(const v16h*)(bbase + kt * 512);
            #pragma unroll
            for (int m = 0; m < 4; ++m) {
                v16h afrag = *(const v16h*)(ldsA + (m * 8 + kt) * 512 + lane * 16);
                acc[m] = __builtin_amdgcn_wmma_f32_16x16x32_f16(
                    false, afrag, false, bfrag, (short)0, acc[m], false, false);
            }
        }
        __syncthreads();   // ldsA consumed; safe to rebuild / reuse as ldsP
    }

    // ---- scatter pre-activations to LDS: ldsP[gate][row 0..63][jl 0..31] ----
    {
        int jl    = ((w & 1) << 4) + (lane & 15);
        int rbase = (lane >> 4) << 3;                    // C/D layout: lanes>=16 hold M=p+8
        #pragma unroll
        for (int m = 0; m < 4; ++m)
            #pragma unroll
            for (int p = 0; p < 8; ++p)
                ldsP[gate][m * 16 + rbase + p][jl] = acc[m][p];
    }
    __syncthreads();

    // ---- pointwise LSTM cell update for this WG's 32 h-columns ----
    for (int e = tid; e < BB * 32; e += 256) {
        int b  = e >> 5;
        int jl = e & 31;
        int j  = wg * 32 + jl;
        float gi = ldsP[0][b][jl] + biascat[        j];
        float gj = ldsP[1][b][jl] + biascat[ 512 +  j];
        float gf = ldsP[2][b][jl] + biascat[1024 +  j];
        float go = ldsP[3][b][jl] + biascat[1536 +  j];
        float i_ = 1.0f / (1.0f + __expf(-gi));
        float jj = tanhf(gj);
        float f_ = 1.0f / (1.0f + __expf(-gf));
        float o_ = 1.0f / (1.0f + __expf(-go));
        float cn = f_ * c_src[b * HH + j] + i_ * jj;
        float hn = o_ * tanhf(cn);
        c_dst[b * HH + j] = cn;
        h_out[b * HH + j] = hn;      // next step reads this slice of d_out as h
    }
}

// ---------------------------------------------------------------------------
extern "C" void kernel_launch(void* const* d_in, const int* in_sizes, int n_in,
                              void* d_out, int out_size, void* d_ws, size_t ws_size,
                              hipStream_t stream) {
    const float* inputs = (const float*)d_in[0];
    const float* hidden = (const float*)d_in[1];
    const float* c0     = (const float*)d_in[2];
    Ptrs P;
    P.wx[0] = (const float*)d_in[3];  P.bx[0] = (const float*)d_in[4];
    P.wx[1] = (const float*)d_in[5];  P.bx[1] = (const float*)d_in[6];
    P.wx[2] = (const float*)d_in[7];  P.bx[2] = (const float*)d_in[8];
    P.wx[3] = (const float*)d_in[9];  P.bx[3] = (const float*)d_in[10];
    P.wh[0] = (const float*)d_in[11]; P.bh[0] = (const float*)d_in[12];
    P.wh[1] = (const float*)d_in[13]; P.bh[1] = (const float*)d_in[14];
    P.wh[2] = (const float*)d_in[15]; P.bh[2] = (const float*)d_in[16];
    P.wh[3] = (const float*)d_in[17]; P.bh[3] = (const float*)d_in[18];

    char* ws = (char*)d_ws;
    _Float16* Bp      = (_Float16*)ws;                                  // 3 MB
    float*    biascat = (float*)(ws + (size_t)KTOT * NCOLS * 2);        // 8 KB
    float*    cbuf    = (float*)(ws + (size_t)KTOT * NCOLS * 2 + NCOLS * 4); // 128 KB
    float*    out     = (float*)d_out;

    // pack weights + biases (runs every call; deterministic)
    prep_pack<<<(KTOT * NCOLS) / 256, 256, 0, stream>>>(P, Bp, biascat);

    const size_t outStride = (size_t)BB * HH;
    for (int t = 0; t < TSTEPS; ++t) {
        const float* x_t   = inputs + (size_t)t * BB * INS;
        const float* h_src = (t == 0) ? hidden : out + (size_t)(t - 1) * outStride;
        const float* c_src = (t == 0) ? c0 : cbuf;
        float* c_dst = (t == TSTEPS - 1)
                           ? out + (size_t)TSTEPS * outStride + outStride  // c_final tail
                           : cbuf;
        float* h_out = out + (size_t)t * outStride;
        lstm_step<<<16, 256, 0, stream>>>(x_t, h_src, c_src, c_dst, Bp, biascat, h_out);
    }
    // h_final = outputs[T-1]
    hipMemcpyAsync(out + (size_t)TSTEPS * outStride,
                   out + (size_t)(TSTEPS - 1) * outStride,
                   outStride * sizeof(float), hipMemcpyDeviceToDevice, stream);
}
